// MambaModelWrapper_57208964382981
// MI455X (gfx1250) — matrix operand order repaired and checked by reading, not compiled, not run on hardware
//
#include <hip/hip_runtime.h>

// ---------------------------------------------------------------------------
// Mamba forward for gfx1250 (MI455X): bf16 WMMA GEMMs + channel-parallel scan
// ---------------------------------------------------------------------------

typedef __attribute__((ext_vector_type(16))) __bf16 v16bf;
typedef __attribute__((ext_vector_type(8)))  __bf16 v8bf;
typedef __attribute__((ext_vector_type(8)))  float  v8f;

// Model dims (fixed by reference)
constexpr int BB    = 2;
constexpr int LL    = 2048;
constexpr int DM    = 1024;          // d_model
constexpr int DI    = 2048;          // expand * d_model
constexpr int NS    = 16;            // d_state
constexpr int DTR   = 64;            // dt_rank
constexpr int XP    = DTR + 2 * NS;  // 96
constexpr int TT    = BB * LL;       // 4096 tokens
constexpr int NIN   = 2 * DI;        // 4096 (in_proj rows)

static inline int cdiv(int a, int b) { return (a + b - 1) / b; }

// ---------------------------------------------------------------------------
// helpers
// ---------------------------------------------------------------------------
__device__ __forceinline__ float silu_f(float v) {
    return v / (1.0f + __expf(-v));
}

// A fragment (16x32 bf16, MxK): lane holds row (lane&15); K chunks of 8 at
// k0+(lane>>4)*8 (elems 0..7) and +16 further (elems 8..15).
__device__ __forceinline__ v16bf load_frag_a(const __bf16* __restrict__ base,
                                             int ld, int m0, int k0, int lane) {
    int r  = m0 + (lane & 15);
    int kb = k0 + ((lane >> 4) << 3);
    const __bf16* p = base + (size_t)r * ld + kb;
    v8bf lo = *(const v8bf*)(p);
    v8bf hi = *(const v8bf*)(p + 16);
    return __builtin_shufflevector(lo, hi, 0, 1, 2, 3, 4, 5, 6, 7,
                                   8, 9, 10, 11, 12, 13, 14, 15);
}

// B fragment (32x16 bf16, KxN) from an N x K row-major matrix (NT GEMM):
// lane holds column (lane&15); K chunks of 8 at k0+(lane>>4)*8 (elems 0..7)
// and +16 further (elems 8..15).
__device__ __forceinline__ v16bf load_frag_b(const __bf16* __restrict__ base,
                                             int ld, int n0, int k0, int lane) {
    int c  = n0 + (lane & 15);
    int kb = k0 + ((lane >> 4) << 3);
    const __bf16* p = base + (size_t)c * ld + kb;
    v8bf lo = *(const v8bf*)(p);
    v8bf hi = *(const v8bf*)(p + 16);
    return __builtin_shufflevector(lo, hi, 0, 1, 2, 3, 4, 5, 6, 7,
                                   8, 9, 10, 11, 12, 13, 14, 15);
}

// ---------------------------------------------------------------------------
// Generic NT GEMM: C[M,N] = A[M,K] * B[N,K]^T, A/B bf16 row-major, C fp32.
// One wave computes a (16*MT) x (16*JT) strip: MT*JT accumulators with full
// A/B fragment reuse.  Requires M % (16*MT) == 0, N % (16*JT) == 0,
// K % 32 == 0 — no residual tiles, so no branches around WMMA (EXEC all-1s).
// ---------------------------------------------------------------------------
template <int MT, int JT>
__global__ __launch_bounds__(256)
void gemm_bf16_nt(const __bf16* __restrict__ A, const __bf16* __restrict__ B,
                  float* __restrict__ C, int M, int N, int K) {
    const int lane  = threadIdx.x & 31;
    // scalarize the wave index so tile selection is provably wave-uniform
    const int wib   = __builtin_amdgcn_readfirstlane(threadIdx.x >> 5);
    const int wave  = blockIdx.x * 8 + wib;
    const int ntile = N / (16 * JT);
    const int mtile = M / (16 * MT);
    if (wave >= mtile * ntile) return;           // scalar branch
    const int m0 = (wave / ntile) * (16 * MT);
    const int n0 = (wave % ntile) * (16 * JT);

    const v8f zero = {0.f, 0.f, 0.f, 0.f, 0.f, 0.f, 0.f, 0.f};
    v8f acc[MT][JT];
#pragma unroll
    for (int mi = 0; mi < MT; ++mi)
#pragma unroll
        for (int j = 0; j < JT; ++j) acc[mi][j] = zero;

    for (int k0 = 0; k0 < K; k0 += 32) {
        v16bf af[MT];
#pragma unroll
        for (int mi = 0; mi < MT; ++mi)
            af[mi] = load_frag_a(A, K, m0 + mi * 16, k0, lane);
        v16bf bf[JT];
#pragma unroll
        for (int j = 0; j < JT; ++j)
            bf[j] = load_frag_b(B, K, n0 + j * 16, k0, lane);
#pragma unroll
        for (int mi = 0; mi < MT; ++mi)
#pragma unroll
            for (int j = 0; j < JT; ++j)
                acc[mi][j] = __builtin_amdgcn_wmma_f32_16x16x32_bf16(
                    false, af[mi], false, bf[j], (short)0, acc[mi][j],
                    false, false);
    }

    const int lh  = lane >> 4;
    const int col = lane & 15;
#pragma unroll
    for (int mi = 0; mi < MT; ++mi)
#pragma unroll
        for (int j = 0; j < JT; ++j)
#pragma unroll
            for (int v = 0; v < 8; ++v) {
                int m = m0 + mi * 16 + v + lh * 8;
                C[(size_t)m * N + n0 + j * 16 + col] = acc[mi][j][v];
            }
}

// ---------------------------------------------------------------------------
// Elementwise / gather kernels
// ---------------------------------------------------------------------------
__global__ void f32_to_bf16_kernel(const float* __restrict__ in,
                                   __bf16* __restrict__ out, int n) {
    int i = blockIdx.x * blockDim.x + threadIdx.x;
    if (i < n) out[i] = (__bf16)in[i];
}

__global__ void embed_gather_kernel(const int* __restrict__ ids,
                                    const float* __restrict__ emb,
                                    __bf16* __restrict__ out) {
    int i = blockIdx.x * blockDim.x + threadIdx.x;
    if (i >= TT * DM) return;
    int t = i / DM, j = i - t * DM;
    out[i] = (__bf16)emb[(size_t)ids[t] * DM + j];
}

// causal depthwise conv (width 4) over x = xz[:, 0:DI], then SiLU.
__global__ void conv_silu_kernel(const float* __restrict__ xz,
                                 const float* __restrict__ cw,
                                 const float* __restrict__ cb,
                                 float* __restrict__ xact,
                                 __bf16* __restrict__ xbf) {
    int i = blockIdx.x * blockDim.x + threadIdx.x;
    if (i >= TT * DI) return;
    int d = i % DI;
    int t = i / DI;           // flat (b,l)
    int l = t % LL;
    const float* col = xz + (size_t)(t - l) * NIN + d;  // row (b,0), col d
    float acc = cb[d];
#pragma unroll
    for (int k = 0; k < 4; ++k) {
        int ls = l - 3 + k;
        if (ls >= 0) acc += cw[d * 4 + k] * col[(size_t)ls * NIN];
    }
    float s = silu_f(acc);
    xact[i] = s;
    xbf[i]  = (__bf16)s;
}

// split dt columns of xdbl (first DTR of XP) into a compact bf16 matrix
__global__ void dt_split_kernel(const float* __restrict__ xdbl,
                                __bf16* __restrict__ dtbf) {
    int i = blockIdx.x * blockDim.x + threadIdx.x;
    if (i >= TT * DTR) return;
    int t = i / DTR, j = i - t * DTR;
    dtbf[i] = (__bf16)xdbl[(size_t)t * XP + j];
}

// delta = softplus(delta_pre + b_dt) in place
__global__ void softplus_kernel(float* __restrict__ delta,
                                const float* __restrict__ b_dt) {
    int i = blockIdx.x * blockDim.x + threadIdx.x;
    if (i >= TT * DI) return;
    float v = delta[i] + b_dt[i % DI];
    delta[i] = (v > 20.0f) ? v : log1pf(__expf(v));
}

// selective scan: one thread per (b, d) channel, 16-state recurrence in
// registers, sequential over L. Overwrites delta[] with y_raw in place.
__global__ __launch_bounds__(256)
void scan_kernel(const float* __restrict__ xact,
                 float* __restrict__ delta,       // in: delta, out: y_raw
                 const float* __restrict__ xdbl,  // B in cols [64,80), C [80,96)
                 const float* __restrict__ A_log) {
    int idx = blockIdx.x * blockDim.x + threadIdx.x;
    if (idx >= BB * DI) return;
    int b = idx / DI;
    int d = idx - b * DI;

    float An[NS];
#pragma unroll
    for (int n = 0; n < NS; ++n) An[n] = -__expf(A_log[d * NS + n]);

    float h[NS];
#pragma unroll
    for (int n = 0; n < NS; ++n) h[n] = 0.0f;

    const size_t tbase = (size_t)b * LL;
    for (int l = 0; l < LL; ++l) {
        size_t ti  = tbase + l;
        size_t ci  = ti * DI + d;
        float u    = xact[ci];
        float dlt  = delta[ci];
        const float* bc = xdbl + ti * XP;
        float du = dlt * u;
        float y  = 0.0f;
#pragma unroll
        for (int n = 0; n < NS; ++n) {
            float dA = __expf(dlt * An[n]);
            h[n] = h[n] * dA + du * bc[DTR + n];       // B_t
            y   += h[n] * bc[DTR + NS + n];            // C_t
        }
        delta[ci] = y;   // reuse buffer as y_raw
    }
}

// y = (y_raw + x*D) * silu(z);  z = xz[:, DI:2*DI].  Output bf16 for out_proj.
__global__ void gate_kernel(const float* __restrict__ yraw,
                            const float* __restrict__ xact,
                            const float* __restrict__ xz,
                            const float* __restrict__ Dvec,
                            __bf16* __restrict__ ybf) {
    int i = blockIdx.x * blockDim.x + threadIdx.x;
    if (i >= TT * DI) return;
    int d = i % DI;
    int t = i / DI;
    float z = xz[(size_t)t * NIN + DI + d];
    float v = (yraw[i] + xact[i] * Dvec[d]) * silu_f(z);
    ybf[i] = (__bf16)v;
}

// ---------------------------------------------------------------------------
// workspace layout (bytes); every size is a multiple of 256
// ---------------------------------------------------------------------------
constexpr size_t SZ_WIN  = (size_t)NIN * DM * 2;   // 8 MB
constexpr size_t SZ_WXP  = (size_t)XP  * DI * 2;   // 384 KB
constexpr size_t SZ_WDT  = (size_t)DI  * DTR * 2;  // 256 KB
constexpr size_t SZ_WOUT = (size_t)DM  * DI * 2;   // 4 MB
constexpr size_t SZ_HBF  = (size_t)TT  * DM * 2;   // 8 MB
constexpr size_t SZ_XZ   = (size_t)TT  * NIN * 4;  // 64 MB
constexpr size_t SZ_XACT = (size_t)TT  * DI * 4;   // 32 MB
constexpr size_t SZ_XBF  = (size_t)TT  * DI * 2;   // 16 MB
constexpr size_t SZ_XDBL = (size_t)TT  * XP * 4;   // 1.5 MB
constexpr size_t SZ_DTBF = (size_t)TT  * DTR * 2;  // 512 KB
constexpr size_t SZ_DELTA= (size_t)TT  * DI * 4;   // 32 MB

constexpr size_t OFF_WIN   = 0;
constexpr size_t OFF_WXP   = OFF_WIN  + SZ_WIN;
constexpr size_t OFF_WDT   = OFF_WXP  + SZ_WXP;
constexpr size_t OFF_WOUT  = OFF_WDT  + SZ_WDT;
constexpr size_t OFF_HBF   = OFF_WOUT + SZ_WOUT;
constexpr size_t OFF_XZ    = OFF_HBF  + SZ_HBF;
constexpr size_t OFF_XACT  = OFF_XZ   + SZ_XZ;
constexpr size_t OFF_XBF   = OFF_XACT + SZ_XACT;
constexpr size_t OFF_XDBL  = OFF_XBF  + SZ_XBF;
constexpr size_t OFF_DTBF  = OFF_XDBL + SZ_XDBL;
constexpr size_t OFF_DELTA = OFF_DTBF + SZ_DTBF;
constexpr size_t OFF_YBF   = OFF_DELTA+ SZ_DELTA;

// ---------------------------------------------------------------------------
extern "C" void kernel_launch(void* const* d_in, const int* in_sizes, int n_in,
                              void* d_out, int out_size, void* d_ws, size_t ws_size,
                              hipStream_t stream) {
    const int*   ids     = (const int*)  d_in[0];
    const float* emb     = (const float*)d_in[1];
    const float* W_in    = (const float*)d_in[2];
    const float* conv_w  = (const float*)d_in[3];
    const float* conv_b  = (const float*)d_in[4];
    const float* W_xproj = (const float*)d_in[5];
    const float* W_dt    = (const float*)d_in[6];
    const float* b_dt    = (const float*)d_in[7];
    const float* A_log   = (const float*)d_in[8];
    const float* Dvec    = (const float*)d_in[9];
    const float* W_out   = (const float*)d_in[10];
    float* out = (float*)d_out;

    char* ws = (char*)d_ws;
    __bf16* win_bf  = (__bf16*)(ws + OFF_WIN);
    __bf16* wxp_bf  = (__bf16*)(ws + OFF_WXP);
    __bf16* wdt_bf  = (__bf16*)(ws + OFF_WDT);
    __bf16* wout_bf = (__bf16*)(ws + OFF_WOUT);
    __bf16* hbf     = (__bf16*)(ws + OFF_HBF);
    float*  xz      = (float*) (ws + OFF_XZ);
    float*  xact    = (float*) (ws + OFF_XACT);
    __bf16* xbf     = (__bf16*)(ws + OFF_XBF);
    float*  xdbl    = (float*) (ws + OFF_XDBL);
    __bf16* dtbf    = (__bf16*)(ws + OFF_DTBF);
    float*  delta   = (float*) (ws + OFF_DELTA);
    __bf16* ybf     = (__bf16*)(ws + OFF_YBF);

    const int THR = 256;
    auto blocks24 = [](int M, int N) {   // MT=2, JT=4 tiling
        return cdiv((M / 32) * (N / 64), 8);
    };
    auto blocks22 = [](int M, int N) {   // MT=2, JT=2 tiling
        return cdiv((M / 32) * (N / 32), 8);
    };

    // weight conversions (fp32 -> bf16)
    f32_to_bf16_kernel<<<cdiv(NIN * DM, THR), THR, 0, stream>>>(W_in,  win_bf,  NIN * DM);
    f32_to_bf16_kernel<<<cdiv(XP  * DI, THR), THR, 0, stream>>>(W_xproj, wxp_bf, XP * DI);
    f32_to_bf16_kernel<<<cdiv(DI * DTR, THR), THR, 0, stream>>>(W_dt,  wdt_bf,  DI * DTR);
    f32_to_bf16_kernel<<<cdiv(DM  * DI, THR), THR, 0, stream>>>(W_out, wout_bf, DM * DI);

    // embedding gather -> bf16 tokens
    embed_gather_kernel<<<cdiv(TT * DM, THR), THR, 0, stream>>>(ids, emb, hbf);

    // in_proj: xz[T,2DI] = h[T,DM] @ W_in[2DI,DM]^T
    gemm_bf16_nt<2, 4><<<blocks24(TT, NIN), THR, 0, stream>>>(hbf, win_bf, xz, TT, NIN, DM);

    // depthwise causal conv + SiLU
    conv_silu_kernel<<<cdiv(TT * DI, THR), THR, 0, stream>>>(xz, conv_w, conv_b, xact, xbf);

    // x_proj: xdbl[T,96] = x[T,DI] @ W_xproj[96,DI]^T   (96 % 32 == 0 -> JT=2)
    gemm_bf16_nt<2, 2><<<blocks22(TT, XP), THR, 0, stream>>>(xbf, wxp_bf, xdbl, TT, XP, DI);

    // dt columns -> compact bf16
    dt_split_kernel<<<cdiv(TT * DTR, THR), THR, 0, stream>>>(xdbl, dtbf);

    // dt_proj: delta_pre[T,DI] = dt[T,DTR] @ W_dt[DI,DTR]^T
    gemm_bf16_nt<2, 4><<<blocks24(TT, DI), THR, 0, stream>>>(dtbf, wdt_bf, delta, TT, DI, DTR);

    // delta = softplus(delta_pre + b_dt)
    softplus_kernel<<<cdiv(TT * DI, THR), THR, 0, stream>>>(delta, b_dt);

    // selective scan (delta buffer becomes y_raw)
    scan_kernel<<<cdiv(BB * DI, THR), THR, 0, stream>>>(xact, delta, xdbl, A_log);

    // skip + gate -> bf16
    gate_kernel<<<cdiv(TT * DI, THR), THR, 0, stream>>>(delta, xact, xz, Dvec, ybf);

    // out_proj: out[T,DM] = y[T,DI] @ W_out[DM,DI]^T
    gemm_bf16_nt<2, 4><<<blocks24(TT, DM), THR, 0, stream>>>(ybf, wout_bf, out, TT, DM, DI);
}
